// Attention_39608188404100
// MI455X (gfx1250) — matrix operand order repaired and checked by reading, not compiled, not run on hardware
//
#include <hip/hip_runtime.h>

// ---------------------------------------------------------------------------
// Types for CDNA5 WMMA (wave32, v_wmma_f32_16x16x32_bf16)
// ---------------------------------------------------------------------------
typedef __bf16 bf16;
typedef __attribute__((ext_vector_type(8)))  bf16  v8bf;
typedef __attribute__((ext_vector_type(16))) bf16  v16bf;
typedef __attribute__((ext_vector_type(8)))  float v8f;
typedef __attribute__((ext_vector_type(4)))  int   v4i;

#define DEV static __device__ __forceinline__

constexpr int Bn   = 64;
constexpr int Nn   = 197;
constexpr int Cn   = 768;
constexpr int Hn   = 12;
constexpr int DHn  = 64;
constexpr int NTOK = Bn * Nn;     // 12608
constexpr int C3   = 3 * Cn;      // 2304
constexpr int NP   = 208;         // padded key length (13 * 16) for aligned rows
constexpr float QSCALE = 0.125f;  // 64^-0.5

// ---------------------------------------------------------------------------
// Async global->LDS DMA (ASYNCcnt path, ISA 08_async_tensor §4), if available
// Builtin signature (from hipcc diagnostic): (AS1 v4i*, AS3 v4i*, imm, imm)
// ---------------------------------------------------------------------------
#if defined(__has_builtin)
#if __has_builtin(__builtin_amdgcn_global_load_async_to_lds_b128)
#define HAVE_ASYNC_LDS 1
#endif
#endif
#ifndef HAVE_ASYNC_LDS
#define HAVE_ASYNC_LDS 0
#endif

#if HAVE_ASYNC_LDS
typedef __attribute__((address_space(1))) v4i* gptr_v4i;
typedef __attribute__((address_space(3))) v4i* lptr_v4i;

DEV void async16(const bf16* g, bf16* l) {
  __builtin_amdgcn_global_load_async_to_lds_b128((gptr_v4i)g, (lptr_v4i)l, 0, 0);
}
DEV void wait_async() {
#if __has_builtin(__builtin_amdgcn_s_wait_asynccnt)
  __builtin_amdgcn_s_wait_asynccnt(0);
#else
  asm volatile("s_wait_asynccnt 0x0" ::: "memory");
#endif
}
#endif

// ---------------------------------------------------------------------------
// Fragment loaders from LDS (16-bit matrices, wave32 layouts; ISA 7.12.2)
// A 16x32 (MxK): lane<16 -> M=lane, K in {0..7,16..23}; lane>=16 -> {8..15,24..31}
// B 32x16 (KxN) stored transposed in LDS as [N][K]:
//   lane<16 -> N=lane, K in {0..7,16..23}; lane>=16 -> {8..15,24..31}
// ---------------------------------------------------------------------------
DEV v16bf load_frag_a(const bf16* row0, int stride, int lane) {
  int r  = lane & 15;
  int kb = (lane & 16) >> 1;              // 0 or 8
  const bf16* p = row0 + r * stride + kb;
  v8bf lo = *(const v8bf*)(p);            // K = kb .. kb+7
  v8bf hi = *(const v8bf*)(p + 16);       // K = kb+16 .. kb+23
  return __builtin_shufflevector(lo, hi, 0,1,2,3,4,5,6,7,8,9,10,11,12,13,14,15);
}

DEV v16bf load_frag_b(const bf16* row0, int stride, int lane) {
  int r  = lane & 15;
  int kb = (lane & 16) >> 1;              // 0 or 8
  const bf16* p = row0 + r * stride + kb;
  v8bf lo = *(const v8bf*)(p);            // K = kb .. kb+7
  v8bf hi = *(const v8bf*)(p + 16);       // K = kb+16 .. kb+23
  return __builtin_shufflevector(lo, hi, 0,1,2,3,4,5,6,7,8,9,10,11,12,13,14,15);
}

DEV v8f wmma_bf16(v16bf a, v16bf b, v8f c) {
  // (neg_a, A, neg_b, B, c_mod, C, reuse_a, reuse_b)
  return __builtin_amdgcn_wmma_f32_16x16x32_bf16(false, a, false, b, (short)0, c,
                                                 false, false);
}

// ---------------------------------------------------------------------------
// Shared 128x128 GEMM core, K = 768, double-buffered LDS pipeline.
// A rows are clamped (OOB rows only feed skipped epilogue rows); B rows are
// always in range by grid construction.
// smem layout: [2 buffers][ A(128x40) | B(128x40) ]
// ---------------------------------------------------------------------------
DEV void gemm_bf16_128x128(const bf16* __restrict__ gA, const bf16* __restrict__ gB,
                           int m0, int n0, int tid, bf16* smem, v8f (&acc)[2][4]) {
  const int lane = tid & 31, wid = tid >> 5;
  const int wm = (wid & 3) * 32, wn = (wid >> 2) * 64;
  constexpr int TILE = 128 * 40;
  constexpr int BUF  = 2 * TILE;
  constexpr int KT   = Cn / 32;           // 24

  auto stage = [&](int kt, int buf) {
    bf16* dA = smem + buf * BUF;
    bf16* dB = dA + TILE;
    const int k0 = kt * 32;
    #pragma unroll
    for (int c = tid; c < 512; c += 256) {
      int r = c >> 2, kk = (c & 3) * 8;
      int gr = m0 + r;  gr = gr < NTOK ? gr : NTOK - 1;   // clamp, no divergence
      int gn = n0 + r;                                    // always valid
#if HAVE_ASYNC_LDS
      async16(gA + (size_t)gr * Cn + k0 + kk, dA + r * 40 + kk);
      async16(gB + (size_t)gn * Cn + k0 + kk, dB + r * 40 + kk);
#else
      *(v8bf*)(dA + r * 40 + kk) = *(const v8bf*)(gA + (size_t)gr * Cn + k0 + kk);
      *(v8bf*)(dB + r * 40 + kk) = *(const v8bf*)(gB + (size_t)gn * Cn + k0 + kk);
#endif
    }
  };

  stage(0, 0);
  for (int kt = 0; kt < KT; kt++) {
    const int cur = kt & 1;
#if HAVE_ASYNC_LDS
    wait_async();                         // my DMA into buf[cur] is done
#endif
    __syncthreads();                      // everyone's copies done, prev reads done
    if (kt + 1 < KT) stage(kt + 1, cur ^ 1);

    const bf16* sA = smem + cur * BUF;
    const bf16* sB = sA + TILE;
    v16bf aF[2], bF[4];
    aF[0] = load_frag_a(sA + wm * 40,        40, lane);
    aF[1] = load_frag_a(sA + (wm + 16) * 40, 40, lane);
    #pragma unroll
    for (int j = 0; j < 4; j++)
      bF[j] = load_frag_b(sB + (wn + j * 16) * 40, 40, lane);
    #pragma unroll
    for (int i = 0; i < 2; i++)
      #pragma unroll
      for (int j = 0; j < 4; j++)
        acc[i][j] = wmma_bf16(aF[i], bF[j], acc[i][j]);
  }
}

// ---------------------------------------------------------------------------
// K1: fp32 -> bf16 conversion
// ---------------------------------------------------------------------------
__global__ void k_cvt(const float* __restrict__ s, bf16* __restrict__ d, int n) {
  int i = blockIdx.x * blockDim.x + threadIdx.x;
  int stride = gridDim.x * blockDim.x;
  for (; i < n; i += stride) d[i] = (bf16)s[i];
}

// ---------------------------------------------------------------------------
// K2: QKV GEMM [NTOK,C] x [C,3C] -> q,k,v bf16 [B,H,N,DH], fused bias+SSF
// ---------------------------------------------------------------------------
__global__ __launch_bounds__(256) void k_qkv(
    const bf16* __restrict__ X, const bf16* __restrict__ W,
    const float* __restrict__ q_bias, const float* __restrict__ v_bias,
    const float* __restrict__ sscale, const float* __restrict__ sshift,
    bf16* __restrict__ Q, bf16* __restrict__ Kk, bf16* __restrict__ V) {
  __shared__ bf16 smem[2 * 2 * 128 * 40];
  const int tid = threadIdx.x, lane = tid & 31, wid = tid >> 5;
  const int m0 = blockIdx.y * 128, n0 = blockIdx.x * 128;
  const int wm = (wid & 3) * 32, wn = (wid >> 2) * 64;

  v8f acc[2][4] = {};
  gemm_bf16_128x128(X, W, m0, n0, tid, smem, acc);

  const int rsel = (lane & 16) >> 1;      // +8 rows for upper half-wave
  #pragma unroll
  for (int i = 0; i < 2; i++)
    #pragma unroll
    for (int j = 0; j < 4; j++)
      #pragma unroll
      for (int e = 0; e < 8; e++) {
        int row = m0 + wm + i * 16 + e + rsel;
        int col = n0 + wn + j * 16 + (lane & 15);
        if (row >= NTOK) continue;
        float val = acc[i][j][e];
        float bias = (col < Cn) ? q_bias[col]
                    : (col < 2 * Cn ? 0.f : v_bias[col - 2 * Cn]);
        val = (val + bias) * sscale[col] + sshift[col];
        int which = col / Cn;
        int cc = col - which * Cn;
        int h = cc >> 6, d = cc & 63;
        int b = row / Nn, n = row - b * Nn;
        size_t off = (((size_t)(b * Hn + h)) * Nn + n) * DHn + d;
        if (which == 0)      Q[off]  = (bf16)(val * QSCALE);
        else if (which == 1) Kk[off] = (bf16)val;
        else                 V[off]  = (bf16)val;
      }
}

// ---------------------------------------------------------------------------
// K3: per (b,h,32-query tile): S = q k^T + rel_bias; row softmax; P bf16[.,NP]
// ---------------------------------------------------------------------------
__global__ __launch_bounds__(256) void k_attn(
    const bf16* __restrict__ Q, const bf16* __restrict__ Kk,
    const float* __restrict__ relTab, const int* __restrict__ relIdx,
    bf16* __restrict__ P) {
  __shared__ bf16  sQ[32][72];
  __shared__ bf16  sK[208][72];
  __shared__ float sS[32][208];
  const int tid = threadIdx.x, lane = tid & 31, wid = tid >> 5;
  const int mt = blockIdx.x, h = blockIdx.y, b = blockIdx.z;
  const int nq0 = mt * 32;
  const size_t base = (((size_t)(b * Hn + h)) * Nn) * DHn;

  { // stage q tile: 32 rows x 64, one 16B chunk per thread
    int r = tid >> 3, kk = (tid & 7) * 8;
    int gq = nq0 + r;
    v8bf v = {};
    if (gq < Nn) v = *(const v8bf*)(Q + base + (size_t)gq * DHn + kk);
    *(v8bf*)&sQ[r][kk] = v;
  }
  for (int c = tid; c < 208 * 8; c += 256) {  // full K matrix for this (b,h)
    int r = c >> 3, kk = (c & 7) * 8;
    v8bf v = {};
    if (r < Nn) v = *(const v8bf*)(Kk + base + (size_t)r * DHn + kk);
    *(v8bf*)&sK[r][kk] = v;
  }
  __syncthreads();

  const int jt[2] = {wid, wid + 8};           // 13 N-tiles over 8 waves
  v8f acc[2][2] = {};
  #pragma unroll
  for (int kc = 0; kc < 2; kc++) {
    v16bf aF0 = load_frag_a(&sQ[0][kc * 32],  72, lane);
    v16bf aF1 = load_frag_a(&sQ[16][kc * 32], 72, lane);
    #pragma unroll
    for (int u = 0; u < 2; u++) {
      if (jt[u] < 13) {                       // wave-uniform branch, EXEC stays full
        v16bf bF = load_frag_b(&sK[jt[u] * 16][kc * 32], 72, lane);
        acc[0][u] = wmma_bf16(aF0, bF, acc[0][u]);
        acc[1][u] = wmma_bf16(aF1, bF, acc[1][u]);
      }
    }
  }

  const int rsel = (lane & 16) >> 1;
  #pragma unroll
  for (int u = 0; u < 2; u++)
    if (jt[u] < 13)
      #pragma unroll
      for (int i = 0; i < 2; i++)
        #pragma unroll
        for (int e = 0; e < 8; e++) {
          int lr = i * 16 + e + rsel;
          int lc = jt[u] * 16 + (lane & 15);
          float s = acc[i][u][e];
          int gq = nq0 + lr;
          if (gq < Nn && lc < Nn)
            s += relTab[relIdx[gq * Nn + lc] * Hn + h];
          sS[lr][lc] = s;
        }
  __syncthreads();

  // row softmax: 8 lanes per row, wave32 shfl_xor reductions
  int r = tid >> 3, p = tid & 7;
  int l0 = p * 25;
  int l1 = (l0 + 25 < Nn) ? l0 + 25 : Nn;
  float mx = -3.0e38f;
  for (int l = l0; l < l1; l++) mx = fmaxf(mx, sS[r][l]);
  #pragma unroll
  for (int m = 1; m < 8; m <<= 1) mx = fmaxf(mx, __shfl_xor(mx, m, 32));
  float sum = 0.f;
  for (int l = l0; l < l1; l++) {
    float e = __expf(sS[r][l] - mx);
    sS[r][l] = e;
    sum += e;
  }
  #pragma unroll
  for (int m = 1; m < 8; m <<= 1) sum += __shfl_xor(sum, m, 32);
  float inv = 1.f / sum;
  int gq = nq0 + r;
  if (gq < Nn) {
    bf16* prow = P + (((size_t)(b * Hn + h)) * Nn + gq) * NP;
    for (int l = l0; l < l1; l++) prow[l] = (bf16)(sS[r][l] * inv);
    if (p == 7)
      for (int l = Nn; l < NP; l++) prow[l] = (bf16)0.f;  // zero pad cols
  }
}

// ---------------------------------------------------------------------------
// K4: DCF head mix  P2[b,k,:,:] = sum_h P[b,h,:,:] * (coeff[k,h] + (h==k))
// ---------------------------------------------------------------------------
__global__ __launch_bounds__(256) void k_mix(
    const bf16* __restrict__ P, const float* __restrict__ coeff,
    bf16* __restrict__ P2) {
  __shared__ float sMix[12][12];
  int tid = threadIdx.x;
  if (tid < 144) {
    int k = tid / 12, hh = tid - k * 12;
    sMix[hh][k] = coeff[k * 12 + hh] + ((hh == k) ? 1.f : 0.f);
  }
  __syncthreads();
  const int NNP = Nn * NP;                    // 40976
  long long idx = (long long)blockIdx.x * 256 + tid;
  if (idx >= (long long)Bn * NNP) return;
  int b = (int)(idx / NNP);
  int rem = (int)(idx - (long long)b * NNP);
  size_t base = (size_t)b * Hn * NNP + rem;
  float pv[12];
  #pragma unroll
  for (int hh = 0; hh < 12; hh++) pv[hh] = (float)P[base + (size_t)hh * NNP];
  #pragma unroll
  for (int k = 0; k < 12; k++) {
    float o = 0.f;
    #pragma unroll
    for (int hh = 0; hh < 12; hh++) o += pv[hh] * sMix[hh][k];
    P2[base + (size_t)k * NNP] = (bf16)o;
  }
}

// ---------------------------------------------------------------------------
// K5: O = P2 @ v per (b,h): [197,197] x [197,64]; v transposed via LDS
// ---------------------------------------------------------------------------
__global__ __launch_bounds__(256) void k_av(
    const bf16* __restrict__ P2, const bf16* __restrict__ V,
    bf16* __restrict__ Yo /* [NTOK, C] */) {
  __shared__ bf16 sP[32][40];
  __shared__ bf16 sV[64][40];                 // v^T tile: [d][l]
  const int tid = threadIdx.x, lane = tid & 31, wid = tid >> 5;
  const int mt = blockIdx.x, h = blockIdx.y, b = blockIdx.z;
  const int n0 = mt * 32;
  const int ti = (wid >> 2) * 16, tj = (wid & 3) * 16;
  const size_t baseP = (((size_t)(b * Hn + h)) * Nn) * NP;
  const size_t baseV = (((size_t)(b * Hn + h)) * Nn) * DHn;

  v8f acc = {};
  for (int lc = 0; lc < 7; lc++) {
    int l0 = lc * 32;
    if (tid < 128) {                          // stage P tile 32x32
      int r = tid >> 2, kk = (tid & 3) * 8;
      int gq = n0 + r, gl = l0 + kk;
      v8bf v = {};
      if (gq < Nn && gl < NP)
        v = *(const v8bf*)(P2 + baseP + (size_t)gq * NP + gl);
      *(v8bf*)&sP[r][kk] = v;
    }
    {                                         // stage v^T tile 64(d) x 32(l)
      int r = tid >> 3, kk = (tid & 7) * 8;   // r = l_local, kk = d offset
      int gl = l0 + r;
      v8bf v = {};
      if (gl < Nn) v = *(const v8bf*)(V + baseV + (size_t)gl * DHn + kk);
      #pragma unroll
      for (int e = 0; e < 8; e++) sV[kk + e][r] = v[e];
    }
    __syncthreads();
    v16bf aF = load_frag_a(&sP[ti][0], 40, lane);
    v16bf bF = load_frag_b(&sV[tj][0], 40, lane);
    acc = wmma_bf16(aF, bF, acc);
    __syncthreads();
  }

  const int rsel = (lane & 16) >> 1;
  #pragma unroll
  for (int e = 0; e < 8; e++) {
    int gq = n0 + ti + e + rsel;
    int d  = tj + (lane & 15);
    if (gq < Nn)
      Yo[((size_t)(b * Nn + gq)) * Cn + h * DHn + d] = (bf16)acc[e];
  }
}

// ---------------------------------------------------------------------------
// K6: output projection  out = Y @ Wp^T + b, SSF, fp32 out [NTOK, C]
// ---------------------------------------------------------------------------
__global__ __launch_bounds__(256) void k_proj(
    const bf16* __restrict__ Y, const bf16* __restrict__ W,
    const float* __restrict__ pb, const float* __restrict__ sscale,
    const float* __restrict__ sshift, float* __restrict__ out) {
  __shared__ bf16 smem[2 * 2 * 128 * 40];
  const int tid = threadIdx.x, lane = tid & 31, wid = tid >> 5;
  const int m0 = blockIdx.y * 128, n0 = blockIdx.x * 128;
  const int wm = (wid & 3) * 32, wn = (wid >> 2) * 64;

  v8f acc[2][4] = {};
  gemm_bf16_128x128(Y, W, m0, n0, tid, smem, acc);

  const int rsel = (lane & 16) >> 1;
  #pragma unroll
  for (int i = 0; i < 2; i++)
    #pragma unroll
    for (int j = 0; j < 4; j++)
      #pragma unroll
      for (int e = 0; e < 8; e++) {
        int row = m0 + wm + i * 16 + e + rsel;
        int col = n0 + wn + j * 16 + (lane & 15);
        if (row >= NTOK) continue;
        float val = acc[i][j][e] + pb[col];
        val = val * sscale[col] + sshift[col];
        out[(size_t)row * Cn + col] = val;
      }
}

// ---------------------------------------------------------------------------
// Launch
// ---------------------------------------------------------------------------
extern "C" void kernel_launch(void* const* d_in, const int* in_sizes, int n_in,
                              void* d_out, int out_size, void* d_ws, size_t ws_size,
                              hipStream_t stream) {
  (void)in_sizes; (void)n_in; (void)out_size; (void)ws_size;
  const float* x       = (const float*)d_in[0];
  const float* qkv_w   = (const float*)d_in[1];
  const float* q_bias  = (const float*)d_in[2];
  const float* v_bias  = (const float*)d_in[3];
  const float* ss_qkv  = (const float*)d_in[4];
  const float* sh_qkv  = (const float*)d_in[5];
  const float* rel_tab = (const float*)d_in[6];
  const float* coeff   = (const float*)d_in[7];
  const float* proj_w  = (const float*)d_in[8];
  const float* proj_b  = (const float*)d_in[9];
  const float* ss_proj = (const float*)d_in[10];
  const float* sh_proj = (const float*)d_in[11];
  const int*   rel_idx = (const int*)d_in[12];
  float* out = (float*)d_out;

  char* ws = (char*)d_ws;
  size_t off = 0;
  auto alloc = [&](size_t bytes) {
    char* p = ws + off;
    off = (off + bytes + 255) & ~(size_t)255;
    return p;
  };
  const size_t szHead = (size_t)Bn * Hn * Nn * DHn * sizeof(bf16);
  const size_t szP    = (size_t)Bn * Hn * Nn * NP  * sizeof(bf16);
  bf16* Xbf = (bf16*)alloc((size_t)NTOK * Cn * sizeof(bf16));
  bf16* Wq  = (bf16*)alloc((size_t)C3 * Cn * sizeof(bf16));
  bf16* Wp  = (bf16*)alloc((size_t)Cn * Cn * sizeof(bf16));
  bf16* Q   = (bf16*)alloc(szHead);
  bf16* Kk  = (bf16*)alloc(szHead);
  bf16* V   = (bf16*)alloc(szHead);
  bf16* P   = (bf16*)alloc(szP);
  bf16* P2  = (bf16*)alloc(szP);

  k_cvt<<<256, 256, 0, stream>>>(x,      Xbf, NTOK * Cn);
  k_cvt<<<128, 256, 0, stream>>>(qkv_w,  Wq,  C3 * Cn);
  k_cvt<<<64,  256, 0, stream>>>(proj_w, Wp,  Cn * Cn);

  k_qkv<<<dim3(C3 / 128, (NTOK + 127) / 128), 256, 0, stream>>>(
      Xbf, Wq, q_bias, v_bias, ss_qkv, sh_qkv, Q, Kk, V);

  k_attn<<<dim3(7, Hn, Bn), 256, 0, stream>>>(Q, Kk, rel_tab, rel_idx, P);

  {
    long long total = (long long)Bn * Nn * NP;
    int blocks = (int)((total + 255) / 256);
    k_mix<<<blocks, 256, 0, stream>>>(P, coeff, P2);
  }

  bf16* Yo = Xbf;  // x is dead after k_qkv: reuse region for attention output
  k_av<<<dim3(7, Hn, Bn), 256, 0, stream>>>(P2, V, Yo);

  k_proj<<<dim3(Cn / 128, (NTOK + 127) / 128), 256, 0, stream>>>(
      Yo, Wp, proj_b, ss_proj, sh_proj, out);
}
